// MultiHeadAttention_7576322310208
// MI455X (gfx1250) — matrix operand order repaired and checked by reading, not compiled, not run on hardware
//
#include <hip/hip_runtime.h>
#include <hip/hip_bf16.h>
#include <stdint.h>

// ---------------------------------------------------------------------------
// MHA forward for MI455X (gfx1250, wave32, WMMA bf16 16x16x32 + TDM)
// B=4, S=2048, D=1024, H=16, DK=64
// ---------------------------------------------------------------------------

typedef __attribute__((ext_vector_type(16))) __bf16 bf16x16;
typedef __attribute__((ext_vector_type(8)))  float  f32x8;
typedef __attribute__((ext_vector_type(4)))  unsigned int u32x4;
typedef __attribute__((ext_vector_type(8)))  int    i32x8;
typedef __attribute__((ext_vector_type(4)))  int    i32x4;

union FragAB {
  bf16x16 v;
  uint4   q[2];
};

__device__ __forceinline__ f32x8 wmma_bf16(const FragAB& a, const FragAB& b, f32x8 c) {
  return __builtin_amdgcn_wmma_f32_16x16x32_bf16(false, a.v, false, b.v, (short)0, c, false, false);
}

// reduce across the two 16-lane halves (lane L <-> L^16): SWAPX16 swizzle
__device__ __forceinline__ float xor16_max(float v) {
  int o = __builtin_amdgcn_ds_swizzle(__float_as_int(v), 0x401F);
  return fmaxf(v, __int_as_float(o));
}
__device__ __forceinline__ float xor16_add(float v) {
  int o = __builtin_amdgcn_ds_swizzle(__float_as_int(v), 0x401F);
  return v + __int_as_float(o);
}
__device__ __forceinline__ float lane_bcast(float v, int srclane) {
  int o = __builtin_amdgcn_ds_bpermute(srclane << 2, __float_as_int(v));
  return __int_as_float(o);
}

// ---------------------------------------------------------------------------
// Tensor Data Mover: 2D tile load, 128 rows x 64 bf16 (row stride 64 elems in
// memory) into LDS with a 16B pad after every 128B row (=> 72-elem LDS rows).
// D# per CDNA5 ISA 8.3/8.4. Issued by one wave; TDM ignores EXEC.
// This toolchain exposes the 6-arg builtin:
//   (u32x4 g0, i32x8 g1, i32x4 g2, i32x4 g3, i32x8 extra, i32 cpol)
// ---------------------------------------------------------------------------
#if defined(__has_builtin)
#if __has_builtin(__builtin_amdgcn_tensor_load_to_lds)
#define HAVE_TDM 1
#endif
#endif

#ifdef HAVE_TDM
__device__ __forceinline__ void tdm_load_k_tile(const __bf16* gsrc, unsigned lds_off) {
  const unsigned long long ga = (unsigned long long)(uintptr_t)gsrc;
  u32x4 g0;
  g0[0] = 1u;                                               // count=1, user D#
  g0[1] = lds_off;                                          // lds_addr
  g0[2] = (unsigned)ga;                                     // global_addr[31:0]
  g0[3] = (unsigned)((ga >> 32) & 0x1FFFFFFu) | (2u << 30); // addr[56:32] | type=2
  i32x8 g1;
  g1[0] = (1 << 16)      // data_size = 2 bytes
        | (1 << 20)      // pad_enable
        | (4 << 22)      // pad_interval: 32 DWORDs (one 128B row)
        | (3 << 25);     // pad_amount:   4 DWORDs (16B) -> 144B LDS row stride
  g1[1] = (64  << 16);   // tensor_dim0 = 64   (bits 79:48)
  g1[2] = (128 << 16);   // tensor_dim1 = 128  (bits 111:80)
  g1[3] = (64  << 16);   // tile_dim0 = 64     (bits 127:112)
  g1[4] = 128;           // tile_dim1 = 128    (bits 143:128), tile_dim2 = 0
  g1[5] = 64;            // tensor_dim0_stride = 64 (bits 207:160)
  g1[6] = 0;
  g1[7] = 0;
  const i32x4 z4 = {0, 0, 0, 0};                  // groups 2/3 unused (2D tile)
  const i32x8 z8 = {0, 0, 0, 0, 0, 0, 0, 0};
  __builtin_amdgcn_tensor_load_to_lds(g0, g1, z4, z4, z8, 0);
}
#endif

// ---------------------------------------------------------------------------
// GEMM: Out[m,n] = sum_k A[m,k] * W[n,k] + bias[n]
//   M=8192, N=1024, K=1024. Block tile 128x128, 8 waves, wave tile 32x64.
//   LDS double-buffered; next slab loaded to registers while WMMAs run.
// MODE 0: A = f32 activations, Out = bf16 in [B,H,S,DK] head-split layout
// MODE 1: A = bf16 context,    Out = f32 flat [M,N]
// ---------------------------------------------------------------------------
template <int MODE>
__global__ __launch_bounds__(256)
void gemm_kernel(const void* __restrict__ Ap, const float* __restrict__ W,
                 const float* __restrict__ bias, void* __restrict__ Out) {
  __shared__ __align__(16) __bf16 As[2][128][40];  // 80B stride: conflict-free
  __shared__ __align__(16) __bf16 Bs[2][128][40];

  const int tid  = threadIdx.x;
  const int lane = tid & 31;
  const int wave = tid >> 5;
  const int m    = lane & 15;
  const int kh   = lane >> 4;
  const int wtm  = wave >> 1;
  const int wtn  = wave & 1;
  const int m0   = blockIdx.x * 128;
  const int n0   = blockIdx.y * 128;
  const int sr   = tid >> 3;        // staging row (0..31)
  const int sc   = (tid & 7) * 4;   // staging col (0..28)

  f32x8 acc[2][4];
#pragma unroll
  for (int i = 0; i < 2; ++i)
#pragma unroll
    for (int j = 0; j < 4; ++j)
      acc[i][j] = (f32x8){0.f,0.f,0.f,0.f,0.f,0.f,0.f,0.f};

  float4 a4[4], w4[4];
  uint2  a2[4];

  // ---- register-stage loaders / LDS writers ----
  auto load_slab = [&](int kk) {
    if (MODE == 0) {
      const float* A = (const float*)Ap;
#pragma unroll
      for (int p = 0; p < 4; ++p)
        a4[p] = *(const float4*)(A + (size_t)(m0 + p * 32 + sr) * 1024 + kk + sc);
    } else {
      const __bf16* A = (const __bf16*)Ap;
#pragma unroll
      for (int p = 0; p < 4; ++p)
        a2[p] = *(const uint2*)(A + (size_t)(m0 + p * 32 + sr) * 1024 + kk + sc);
    }
#pragma unroll
    for (int p = 0; p < 4; ++p)
      w4[p] = *(const float4*)(W + (size_t)(n0 + p * 32 + sr) * 1024 + kk + sc);
  };
  auto store_slab = [&](int buf) {
    if (MODE == 0) {
#pragma unroll
      for (int p = 0; p < 4; ++p) {
        union { __bf16 h[4]; uint2 u; } pk;
        pk.h[0] = (__bf16)a4[p].x; pk.h[1] = (__bf16)a4[p].y;
        pk.h[2] = (__bf16)a4[p].z; pk.h[3] = (__bf16)a4[p].w;
        *(uint2*)&As[buf][p * 32 + sr][sc] = pk.u;
      }
    } else {
#pragma unroll
      for (int p = 0; p < 4; ++p)
        *(uint2*)&As[buf][p * 32 + sr][sc] = a2[p];
    }
#pragma unroll
    for (int p = 0; p < 4; ++p) {
      union { __bf16 h[4]; uint2 u; } pk;
      pk.h[0] = (__bf16)w4[p].x; pk.h[1] = (__bf16)w4[p].y;
      pk.h[2] = (__bf16)w4[p].z; pk.h[3] = (__bf16)w4[p].w;
      *(uint2*)&Bs[buf][p * 32 + sr][sc] = pk.u;
    }
  };

  // ---- prologue: fill buffer 0 ----
  load_slab(0);
  store_slab(0);
  __syncthreads();

  // ---- pipelined main loop: 32 K-slabs ----
  for (int it = 0; it < 32; ++it) {
    const int cur = it & 1;
    const bool has_next = (it + 1) < 32;
    if (has_next) load_slab((it + 1) * 32);   // HBM loads overlap WMMAs below

    FragAB af[2], bfr[4];
#pragma unroll
    for (int i = 0; i < 2; ++i) {
      const __bf16* p = &As[cur][wtm * 32 + i * 16 + m][0];
      af[i].q[0] = *(const uint4*)(p + kh * 16);
      af[i].q[1] = *(const uint4*)(p + kh * 16 + 8);
    }
#pragma unroll
    for (int j = 0; j < 4; ++j) {
      const __bf16* p = &Bs[cur][wtn * 64 + j * 16 + m][0];
      bfr[j].q[0] = *(const uint4*)(p + kh * 16);
      bfr[j].q[1] = *(const uint4*)(p + kh * 16 + 8);
    }
#pragma unroll
    for (int i = 0; i < 2; ++i)
#pragma unroll
      for (int j = 0; j < 4; ++j)
        acc[i][j] = wmma_bf16(af[i], bfr[j], acc[i][j]);

    if (has_next) store_slab(cur ^ 1);
    __syncthreads();
  }

  // ---- epilogue ----
#pragma unroll
  for (int i = 0; i < 2; ++i) {
#pragma unroll
    for (int j = 0; j < 4; ++j) {
      const int gmb = m0 + wtm * 32 + i * 16 + kh * 8;
      const int gn  = n0 + wtn * 64 + j * 16 + m;
      const float bv = bias[gn];
      if (MODE == 1) {
        float* O = (float*)Out;
#pragma unroll
        for (int r = 0; r < 8; ++r)
          O[(size_t)(gmb + r) * 1024 + gn] = acc[i][j][r] + bv;
      } else {
        __bf16* O = (__bf16*)Out;
        const int hh = gn >> 6, dk = gn & 63;
#pragma unroll
        for (int r = 0; r < 8; ++r) {
          const int gm = gmb + r;
          const int bb = gm >> 11, ss = gm & 2047;
          O[(((size_t)(bb * 16 + hh)) * 2048 + ss) * 64 + dk] =
              (__bf16)(acc[i][j][r] + bv);
        }
      }
    }
  }
}

// ---------------------------------------------------------------------------
// Flash attention: one (b,h) x 128 q-rows per block; wave owns 16 q-rows.
// K tile staged by the Tensor Data Mover (wave 0 issues, TENSORcnt fence);
// V tile staged transposed via 4x4 register micro-transposes (ds_store_b64).
// S^T = K·Q^T so softmax stats are lane-resident and exp(S) converts
// in-register into the P A-fragment layout for P·V.
// ---------------------------------------------------------------------------
__global__ __launch_bounds__(256)
void attn_kernel(const __bf16* __restrict__ Q, const __bf16* __restrict__ K,
                 const __bf16* __restrict__ V, __bf16* __restrict__ Ctx) {
  __shared__ __align__(16) __bf16 Ks[128][72];   // [key][d], 144B stride
  __shared__ __align__(16) __bf16 Vts[64][136];  // [d][key], 272B stride

  const int tid  = threadIdx.x;
  const int lane = tid & 31;
  const int wave = tid >> 5;
  const int nidx = lane & 15;
  const int kh   = lane >> 4;
  const int bh   = blockIdx.y;                   // b*16 + h
  const size_t base = (size_t)bh * 2048 * 64;
  const int q0   = blockIdx.x * 128 + wave * 16;

  // Q B-fragments, kept in registers for the whole kernel
  FragAB qf[2];
#pragma unroll
  for (int c = 0; c < 2; ++c) {
    const __bf16* p = Q + base + (size_t)(q0 + nidx) * 64 + c * 32 + kh * 16;
    qf[c].q[0] = *(const uint4*)p;
    qf[c].q[1] = *(const uint4*)(p + 8);
  }

  f32x8 o[4];
#pragma unroll
  for (int n = 0; n < 4; ++n) o[n] = (f32x8){0.f,0.f,0.f,0.f,0.f,0.f,0.f,0.f};
  float m_run = -1e30f, l_run = 0.f;

  const unsigned ks_off = (unsigned)(uintptr_t)&Ks[0][0];
  (void)ks_off;

  for (int kb = 0; kb < 16; ++kb) {
    const int key0 = kb * 128;

    // ---- stage K tile: TDM DMA (wave 0) or vector-copy fallback ----
#ifdef HAVE_TDM
    if (wave == 0) tdm_load_k_tile(K + base + (size_t)key0 * 64, ks_off);
#else
#pragma unroll
    for (int p = 0; p < 4; ++p) {
      const int r = p * 32 + (tid >> 3);
      const int c = (tid & 7) * 8;
      *(uint4*)&Ks[r][c] = *(const uint4*)(K + base + (size_t)(key0 + r) * 64 + c);
    }
#endif

    // ---- stage V transposed: Vts[d][key], 4x4 register micro-transpose ----
#pragma unroll
    for (int p = 0; p < 2; ++p) {
      const int bid = p * 256 + tid;
      const int db4 = (bid & 15) * 4;   // d block (coalesced global reads)
      const int kb4 = (bid >> 4) * 4;   // key block
      union { uint2 u; __bf16 h[4]; } row[4], out;
#pragma unroll
      for (int i = 0; i < 4; ++i)
        row[i].u = *(const uint2*)(V + base + (size_t)(key0 + kb4 + i) * 64 + db4);
#pragma unroll
      for (int e = 0; e < 4; ++e) {
#pragma unroll
        for (int i = 0; i < 4; ++i) out.h[i] = row[i].h[e];
        *(uint2*)&Vts[db4 + e][kb4] = out.u;
      }
    }

#ifdef HAVE_TDM
    if (wave == 0) __builtin_amdgcn_s_wait_tensorcnt(0);
#endif
    __syncthreads();

    // ---- S^T tiles: 8 tiles of (16 keys x 16 q) ----
    f32x8 st[8];
#pragma unroll
    for (int t = 0; t < 8; ++t) {
      f32x8 a = (f32x8){0.f,0.f,0.f,0.f,0.f,0.f,0.f,0.f};
#pragma unroll
      for (int c = 0; c < 2; ++c) {
        FragAB kf;
        const __bf16* p = &Ks[t * 16 + nidx][c * 32 + kh * 16];
        kf.q[0] = *(const uint4*)p;
        kf.q[1] = *(const uint4*)(p + 8);
        a = wmma_bf16(kf, qf[c], a);
      }
      st[t] = a;
    }

    // ---- online softmax (lane = q) ----
    float pmax = -1e30f;
#pragma unroll
    for (int t = 0; t < 8; ++t)
#pragma unroll
      for (int r = 0; r < 8; ++r) {
        st[t][r] *= 0.125f;  // 1/sqrt(64)
        pmax = fmaxf(pmax, st[t][r]);
      }
    pmax = xor16_max(pmax);
    const float m_new = fmaxf(m_run, pmax);
    const float rescale = __expf(m_run - m_new);
    m_run = m_new;

    float rsum = 0.f;
#pragma unroll
    for (int t = 0; t < 8; ++t)
#pragma unroll
      for (int r = 0; r < 8; ++r) {
        const float pv = __expf(st[t][r] - m_new);
        st[t][r] = pv;
        rsum += pv;
      }
    rsum = xor16_add(rsum);
    l_run = l_run * rescale + rsum;

    float rr[8];
#pragma unroll
    for (int r = 0; r < 8; ++r) rr[r] = lane_bcast(rescale, kh * 8 + r);
#pragma unroll
    for (int n = 0; n < 4; ++n)
#pragma unroll
      for (int r = 0; r < 8; ++r) o[n][r] *= rr[r];

    // ---- P·V ----
#pragma unroll
    for (int kc = 0; kc < 4; ++kc) {
      FragAB pf;
#pragma unroll
      for (int e = 0; e < 8; ++e) {
        pf.v[e]     = (__bf16)st[2 * kc][e];
        pf.v[8 + e] = (__bf16)st[2 * kc + 1][e];
      }
#pragma unroll
      for (int n = 0; n < 4; ++n) {
        FragAB vf;
        const __bf16* p = &Vts[n * 16 + nidx][kc * 32 + kh * 8];
        vf.q[0] = *(const uint4*)p;
        vf.q[1] = *(const uint4*)(p + 16);
        o[n] = wmma_bf16(pf, vf, o[n]);
      }
    }
    __syncthreads();
  }

  // ---- epilogue ----
  const float linv = 1.0f / l_run;
  float li[8];
#pragma unroll
  for (int r = 0; r < 8; ++r) li[r] = lane_bcast(linv, kh * 8 + r);
  const int bb = bh >> 4, hh = bh & 15;
#pragma unroll
  for (int n = 0; n < 4; ++n) {
    const int d = n * 16 + nidx;
#pragma unroll
    for (int r = 0; r < 8; ++r) {
      const int s = q0 + kh * 8 + r;
      Ctx[((size_t)(bb * 2048 + s)) * 1024 + hh * 64 + d] =
          (__bf16)(o[n][r] * li[r]);
    }
  }
}

// ---------------------------------------------------------------------------
extern "C" void kernel_launch(void* const* d_in, const int* in_sizes, int n_in,
                              void* d_out, int out_size, void* d_ws, size_t ws_size,
                              hipStream_t stream) {
  (void)in_sizes; (void)n_in; (void)out_size; (void)ws_size;
  const float* q  = (const float*)d_in[0];
  const float* k  = (const float*)d_in[1];
  const float* v  = (const float*)d_in[2];
  const float* wq = (const float*)d_in[3];
  const float* bq = (const float*)d_in[4];
  const float* wk = (const float*)d_in[5];
  const float* bk = (const float*)d_in[6];
  const float* wv = (const float*)d_in[7];
  const float* bv = (const float*)d_in[8];
  const float* wo = (const float*)d_in[9];
  const float* bo = (const float*)d_in[10];

  const size_t NE = (size_t)4 * 2048 * 1024;  // 8,388,608 elements per tensor
  __bf16* Qp = (__bf16*)d_ws;
  __bf16* Kp = Qp + NE;
  __bf16* Vp = Kp + NE;
  __bf16* Cp = Vp + NE;

  dim3 blk(256);
  dim3 gGemm(64, 8);   // 8192/128 x 1024/128
  dim3 gAttn(16, 64);  // S/128 x B*H

  gemm_kernel<0><<<gGemm, blk, 0, stream>>>(q, wq, bq, Qp);
  gemm_kernel<0><<<gGemm, blk, 0, stream>>>(k, wk, bk, Kp);
  gemm_kernel<0><<<gGemm, blk, 0, stream>>>(v, wv, bv, Vp);
  attn_kernel<<<gAttn, blk, 0, stream>>>(Qp, Kp, Vp, Cp);
  gemm_kernel<1><<<gGemm, blk, 0, stream>>>(Cp, wo, bo, (float*)d_out);
}
